// Block_18064632447630
// MI455X (gfx1250) — matrix operand order repaired and checked by reading, not compiled, not run on hardware
//
#include <hip/hip_runtime.h>
#include <hip/hip_bf16.h>

typedef __attribute__((ext_vector_type(16))) __bf16 v16bf;
typedef __attribute__((ext_vector_type(8)))  float  v8f;

#define BM 128
#define BN 128
#define BKK 32

// ---------------------------------------------------------------------------
// Tiled GEMM: Y[M,N] = A[M,K] @ W[K,N] + bias, optional ReLU.
// A,W fp32 in global; converted to bf16 while staging to LDS; WMMA bf16 f32-acc.
// Requires: M%128==0, N%128==0, K%32==0. 256 threads = 8 waves; wave does 64x32.
// ---------------------------------------------------------------------------
__global__ void __launch_bounds__(256)
gemm_bf16_kernel(const float* __restrict__ A, const float* __restrict__ W,
                 const float* __restrict__ bias, float* __restrict__ Y,
                 int M, int N, int K, int relu)
{
    __shared__ __bf16 As[BM * BKK];   // [m][k], m-major
    __shared__ __bf16 Bs[BN * BKK];   // [n][k], n-major (transposed from W)

    const int tid  = threadIdx.x;
    const int wave = tid >> 5;
    const int lane = tid & 31;
    const int wm   = wave >> 2;        // 0..1  -> 64 rows of M
    const int wn   = wave & 3;         // 0..3  -> 32 cols of N
    const int hlf  = lane >> 4;        // lane half (0/1)
    const int l16  = lane & 15;

    const int tileM = blockIdx.y * BM;
    const int tileN = blockIdx.x * BN;

    v8f acc[4][2];
    #pragma unroll
    for (int i = 0; i < 4; ++i)
        #pragma unroll
        for (int j = 0; j < 2; ++j)
            #pragma unroll
            for (int r = 0; r < 8; ++r) acc[i][j][r] = 0.0f;

    const int nk = K / BKK;
    for (int kt = 0; kt < nk; ++kt) {
        // ---- stage A tile (128x32 fp32 -> bf16), 1024 float4 over 256 thr ----
        #pragma unroll
        for (int r = 0; r < 4; ++r) {
            const int f4 = tid * 4 + r;           // 0..1023
            const int m  = f4 >> 3;               // 0..127
            const int kc = (f4 & 7) << 2;         // 0..28
            const float4 av = *(const float4*)(A + (size_t)(tileM + m) * K + kt * BKK + kc);
            __bf16* p = &As[m * BKK + kc];
            p[0] = (__bf16)av.x; p[1] = (__bf16)av.y;
            p[2] = (__bf16)av.z; p[3] = (__bf16)av.w;
        }
        // ---- stage B tile (W[k][n] 32x128 -> Bs[n][k] bf16) ----
        #pragma unroll
        for (int r = 0; r < 4; ++r) {
            const int f4 = tid * 4 + r;
            const int k  = f4 >> 5;               // 0..31
            const int nc = (f4 & 31) << 2;        // 0..124
            const float4 bv = *(const float4*)(W + (size_t)(kt * BKK + k) * N + tileN + nc);
            Bs[(nc + 0) * BKK + k] = (__bf16)bv.x;
            Bs[(nc + 1) * BKK + k] = (__bf16)bv.y;
            Bs[(nc + 2) * BKK + k] = (__bf16)bv.z;
            Bs[(nc + 3) * BKK + k] = (__bf16)bv.w;
        }
        __syncthreads();

        // ---- build fragments per ISA 16-bit A(16x32)/B(32x16) layouts ----
        v16bf afrag[4], bfrag[2];
        #pragma unroll
        for (int i = 0; i < 4; ++i) {
            const int m = wm * 64 + i * 16 + l16;
            const __bf16* base = &As[m * BKK];
            #pragma unroll
            for (int e = 0; e < 16; ++e) {
                const int j  = e >> 1;
                const int kk = ((j < 4) ? 0 : 16) + hlf * 8 + (j & 3) * 2 + (e & 1);
                afrag[i][e] = base[kk];
            }
        }
        #pragma unroll
        for (int j = 0; j < 2; ++j) {
            const int n = wn * 32 + j * 16 + l16;
            const __bf16* base = &Bs[n * BKK];
            #pragma unroll
            for (int e = 0; e < 16; ++e) bfrag[j][e] = base[((e < 8) ? 0 : 16) + hlf * 8 + (e & 7)];  // k = (e<8?0:16) + hlf*8 + (e&7)
        }

        #pragma unroll
        for (int i = 0; i < 4; ++i)
            #pragma unroll
            for (int j = 0; j < 2; ++j)
                acc[i][j] = __builtin_amdgcn_wmma_f32_16x16x32_bf16(
                    false, afrag[i], false, bfrag[j], (short)0, acc[i][j], false, false);
        __syncthreads();
    }

    // ---- epilogue: bias (+ReLU), C layout: VGPR r -> M = r + 8*hlf, N = l16 ----
    #pragma unroll
    for (int i = 0; i < 4; ++i) {
        #pragma unroll
        for (int j = 0; j < 2; ++j) {
            const int col = tileN + wn * 32 + j * 16 + l16;
            const float bsv = bias[col];
            #pragma unroll
            for (int r = 0; r < 8; ++r) {
                const int row = tileM + wm * 64 + i * 16 + hlf * 8 + r;
                float v = acc[i][j][r] + bsv;
                if (relu) v = v > 0.0f ? v : 0.0f;
                Y[(size_t)row * N + col] = v;
            }
        }
    }
}

// ---------------------------------------------------------------------------
// Gather the 2G=128 selected K/V source rows per batch (sel = [0..63, rand]).
// ---------------------------------------------------------------------------
__global__ void __launch_bounds__(256)
gather_rows_kernel(const float* __restrict__ prev, const int* __restrict__ ridx,
                   float* __restrict__ out, int* __restrict__ selOut)
{
    const int row = blockIdx.x;        // 0 .. 511
    const int b = row >> 7;
    const int j = row & 127;
    const int s = (j < 64) ? j : ridx[b * 64 + (j - 64)];
    if (threadIdx.x == 0) selOut[row] = s;
    const float* src = prev + ((size_t)b * 4096 + s) * 768;
    float* dst = out + (size_t)row * 768;
    for (int d = threadIdx.x; d < 768; d += 256) dst[d] = src[d];
}

// ---------------------------------------------------------------------------
// Sparse attention over 128 selected keys, causal-masked by sel index.
// Block = (b, h) x 128-token tile; K/V head slices staged in LDS (64 KB).
// ---------------------------------------------------------------------------
__global__ void __launch_bounds__(128)
sparse_attn_kernel(const float* __restrict__ Q, const float* __restrict__ Ks,
                   const float* __restrict__ Vs, const int* __restrict__ selIdx,
                   float* __restrict__ O)
{
    __shared__ __align__(16) float Kl[128][64];
    __shared__ __align__(16) float Vl[128][64];
    const int bh = blockIdx.x;
    const int b = bh / 12;
    const int h = bh - b * 12;
    const int t = threadIdx.x;
    const int s = blockIdx.y * 128 + t;
    {
        const float* kr = Ks + (size_t)(b * 128 + t) * 768 + h * 64;
        const float* vr = Vs + (size_t)(b * 128 + t) * 768 + h * 64;
        #pragma unroll
        for (int d = 0; d < 64; d += 4) {
            *(float4*)&Kl[t][d] = *(const float4*)&kr[d];
            *(float4*)&Vl[t][d] = *(const float4*)&vr[d];
        }
    }
    __syncthreads();

    float q[64];
    const float* qr = Q + (size_t)(b * 4096 + s) * 768 + h * 64;
    #pragma unroll
    for (int d = 0; d < 64; d += 4) {
        const float4 v = *(const float4*)&qr[d];
        q[d] = v.x; q[d + 1] = v.y; q[d + 2] = v.z; q[d + 3] = v.w;
    }
    const int* selb = selIdx + b * 128;
    const float scale = 0.125f;  // 1/sqrt(64)

    // pass 1: max (sel[0]=0 <= s always, so at least one live key)
    float m = -1.0e9f;
    for (int k = 0; k < 128; ++k) {
        if (selb[k] <= s) {
            float d0 = 0.0f;
            #pragma unroll
            for (int d = 0; d < 64; ++d) d0 += q[d] * Kl[k][d];
            m = fmaxf(m, d0 * scale);
        }
    }
    // pass 2: exp-sum + weighted V accumulation (masked entries exp->0 exactly)
    float l = 0.0f;
    float out[64];
    #pragma unroll
    for (int d = 0; d < 64; ++d) out[d] = 0.0f;
    for (int k = 0; k < 128; ++k) {
        if (selb[k] <= s) {
            float d0 = 0.0f;
            #pragma unroll
            for (int d = 0; d < 64; ++d) d0 += q[d] * Kl[k][d];
            const float p = expf(d0 * scale - m);
            l += p;
            #pragma unroll
            for (int d = 0; d < 64; ++d) out[d] += p * Vl[k][d];
        }
    }
    const float inv = 1.0f / l;
    float* orow = O + (size_t)(b * 4096 + s) * 768 + h * 64;
    #pragma unroll
    for (int d = 0; d < 64; d += 4) {
        float4 v;
        v.x = out[d] * inv; v.y = out[d + 1] * inv;
        v.z = out[d + 2] * inv; v.w = out[d + 3] * inv;
        *(float4*)&orow[d] = v;
    }
}

// ---------------------------------------------------------------------------
// Class attention, collapsed rank-1 form:
//   sc[c] = scale*(qc.Wkc_h)*cls[c] + const  ->  oc = t*Wvc_h + bvc_h,
//   t = sum_c softmax(sc)_c * cls[c].
// ---------------------------------------------------------------------------
__global__ void __launch_bounds__(256)
class_attn_kernel(const float* __restrict__ QC, const float* __restrict__ cls,
                  const float* __restrict__ Wkc, const float* __restrict__ bkc,
                  const float* __restrict__ Wvc, const float* __restrict__ bvc,
                  float* __restrict__ OC)
{
    __shared__ float cl[256];
    __shared__ float wk[768], bk[768], wv[768], bvw[768];
    const int b = blockIdx.x;
    const int t = threadIdx.x;
    cl[t] = cls[b * 256 + t];
    for (int d = t; d < 768; d += 256) {
        wk[d] = Wkc[d]; bk[d] = bkc[d]; wv[d] = Wvc[d]; bvw[d] = bvc[d];
    }
    __syncthreads();

    const int s = blockIdx.y * 256 + t;
    const float* qrow = QC + ((size_t)b * 4096 + s) * 768;
    float* orow = OC + ((size_t)b * 4096 + s) * 768;
    const float scale = 0.125f;
    for (int h = 0; h < 12; ++h) {
        const int d0 = h * 64;
        float a1 = 0.0f;
        #pragma unroll
        for (int d = 0; d < 64; ++d) a1 += qrow[d0 + d] * wk[d0 + d];
        a1 *= scale;  // the bkc-dot is a constant shift; cancels in softmax
        float m = -3.0e38f;
        for (int c = 0; c < 256; ++c) m = fmaxf(m, a1 * cl[c]);
        float l = 0.0f, tn = 0.0f;
        for (int c = 0; c < 256; ++c) {
            const float e = expf(a1 * cl[c] - m);
            l += e; tn += e * cl[c];
        }
        const float tv = tn / l;
        #pragma unroll
        for (int d = 0; d < 64; ++d) orow[d0 + d] = tv * wv[d0 + d] + bvw[d0 + d];
    }
}

// ---------------------------------------------------------------------------
// Y = LayerNorm(X + R) * g + b, one 256-thread block per row of 768.
// ---------------------------------------------------------------------------
__global__ void __launch_bounds__(256)
add_ln_kernel(const float* __restrict__ X, const float* __restrict__ R,
              const float* __restrict__ g, const float* __restrict__ bta,
              float* __restrict__ Y)
{
    __shared__ float red[8];
    const int row = blockIdx.x;
    const int t = threadIdx.x;
    const float* x = X + (size_t)row * 768;
    const float* r = R + (size_t)row * 768;
    const float v0 = x[t] + r[t];
    const float v1 = x[t + 256] + r[t + 256];
    const float v2 = x[t + 512] + r[t + 512];
    float sm = v0 + v1 + v2;
    for (int o = 16; o > 0; o >>= 1) sm += __shfl_down(sm, o, 32);
    if ((t & 31) == 0) red[t >> 5] = sm;
    __syncthreads();
    float tot = 0.0f;
    #pragma unroll
    for (int i = 0; i < 8; ++i) tot += red[i];
    const float mean = tot * (1.0f / 768.0f);
    __syncthreads();
    const float d0 = v0 - mean, d1 = v1 - mean, d2 = v2 - mean;
    float sv = d0 * d0 + d1 * d1 + d2 * d2;
    for (int o = 16; o > 0; o >>= 1) sv += __shfl_down(sv, o, 32);
    if ((t & 31) == 0) red[t >> 5] = sv;
    __syncthreads();
    float vt = 0.0f;
    #pragma unroll
    for (int i = 0; i < 8; ++i) vt += red[i];
    const float rstd = rsqrtf(vt * (1.0f / 768.0f) + 1e-5f);
    float* y = Y + (size_t)row * 768;
    y[t]       = d0 * rstd * g[t]       + bta[t];
    y[t + 256] = d1 * rstd * g[t + 256] + bta[t + 256];
    y[t + 512] = d2 * rstd * g[t + 512] + bta[t + 512];
}

extern "C" void kernel_launch(void* const* d_in, const int* in_sizes, int n_in,
                              void* d_out, int out_size, void* d_ws, size_t ws_size,
                              hipStream_t stream)
{
    const float* cur  = (const float*)d_in[0];
    const float* prev = (const float*)d_in[1];
    const float* cls  = (const float*)d_in[2];
    const int*   ridx = (const int*)d_in[3];
    const float* Wq  = (const float*)d_in[4];  const float* bq  = (const float*)d_in[5];
    const float* Wk  = (const float*)d_in[6];  const float* bk  = (const float*)d_in[7];
    const float* Wv  = (const float*)d_in[8];  const float* bv  = (const float*)d_in[9];
    const float* Wo  = (const float*)d_in[10]; const float* bo  = (const float*)d_in[11];
    const float* g1  = (const float*)d_in[12]; const float* b1  = (const float*)d_in[13];
    const float* Wqc = (const float*)d_in[14]; const float* bqc = (const float*)d_in[15];
    const float* Wkc = (const float*)d_in[16]; const float* bkc = (const float*)d_in[17];
    const float* Wvc = (const float*)d_in[18]; const float* bvc = (const float*)d_in[19];
    const float* Woc = (const float*)d_in[20]; const float* boc = (const float*)d_in[21];
    const float* g2  = (const float*)d_in[22]; const float* b2  = (const float*)d_in[23];
    const float* W1  = (const float*)d_in[24]; const float* bf1 = (const float*)d_in[25];
    const float* W2  = (const float*)d_in[26]; const float* bf2 = (const float*)d_in[27];
    const float* g3  = (const float*)d_in[28]; const float* b3  = (const float*)d_in[29];

    char* ws = (char*)d_ws;
    const size_t MSD = (size_t)16384 * 768 * 4;        // 48 MB
    const size_t FFB = (size_t)16384 * 3072 * 4;       // 192 MB
    float* R0 = (float*)(ws);                  // Q -> qc -> x2
    float* R1 = (float*)(ws + MSD);            // attn_out -> oc
    float* R2 = (float*)(ws + 2 * MSD);        // gemm temp
    float* R3 = (float*)(ws + 3 * MSD);        // x1
    float* R4 = (float*)(ws + 4 * MSD);        // FF hidden (192 MB)
    float* Psel = (float*)(ws + 4 * MSD + FFB);
    float* Ksl  = Psel + (size_t)512 * 768;
    float* Vsl  = Ksl  + (size_t)512 * 768;
    int*   selIdx = (int*)(Vsl + (size_t)512 * 768);

    // 1) gather the 512 K/V source rows + selection indices
    gather_rows_kernel<<<512, 256, 0, stream>>>(prev, ridx, Psel, selIdx);
    // 2) Q = cur @ Wq + bq
    gemm_bf16_kernel<<<dim3(6, 128), 256, 0, stream>>>(cur, Wq, bq, R0, 16384, 768, 768, 0);
    // 3) Ksel / Vsel projections (only 512 rows!)
    gemm_bf16_kernel<<<dim3(6, 4), 256, 0, stream>>>(Psel, Wk, bk, Ksl, 512, 768, 768, 0);
    gemm_bf16_kernel<<<dim3(6, 4), 256, 0, stream>>>(Psel, Wv, bv, Vsl, 512, 768, 768, 0);
    // 4) sparse masked attention
    sparse_attn_kernel<<<dim3(48, 32), 128, 0, stream>>>(R0, Ksl, Vsl, selIdx, R1);
    // 5) output projection + residual LN -> x1
    gemm_bf16_kernel<<<dim3(6, 128), 256, 0, stream>>>(R1, Wo, bo, R2, 16384, 768, 768, 0);
    add_ln_kernel<<<16384, 256, 0, stream>>>(R2, cur, g1, b1, R3);
    // 6) class attention (rank-1 collapsed)
    gemm_bf16_kernel<<<dim3(6, 128), 256, 0, stream>>>(R3, Wqc, bqc, R0, 16384, 768, 768, 0);
    class_attn_kernel<<<dim3(4, 16), 256, 0, stream>>>(R0, cls, Wkc, bkc, Wvc, bvc, R1);
    gemm_bf16_kernel<<<dim3(6, 128), 256, 0, stream>>>(R1, Woc, boc, R2, 16384, 768, 768, 0);
    add_ln_kernel<<<16384, 256, 0, stream>>>(R2, R3, g2, b2, R0);   // x2
    // 7) FFN
    gemm_bf16_kernel<<<dim3(24, 128), 256, 0, stream>>>(R0, W1, bf1, R4, 16384, 3072, 768, 1);
    gemm_bf16_kernel<<<dim3(6, 128), 256, 0, stream>>>(R4, W2, bf2, R2, 16384, 768, 3072, 0);
    add_ln_kernel<<<16384, 256, 0, stream>>>(R2, R0, g3, b3, (float*)d_out);
}
